// Model_52561809768912
// MI455X (gfx1250) — matrix operand order repaired and checked by reading, not compiled, not run on hardware
//
#include <hip/hip_runtime.h>
#include <hip/hip_bf16.h>

// Problem dims
#define Bsz 64
#define Tlen 96
#define Pn 256
#define Hd 128
#define Od 24
#define G4 512   // 4*Hd

typedef __attribute__((ext_vector_type(16))) __bf16 v16bf;
typedef __attribute__((ext_vector_type(8)))  __bf16 v8bf;
typedef __attribute__((ext_vector_type(2)))  __bf16 v2bf;
typedef __attribute__((ext_vector_type(8)))  float  v8f;

// LDS layout (byte offsets):
//   w_lds   : [512][128] bf16   @ 0       (131072)   K-columns stored permuted
//   h_lds   : [64][128]  bf16   @ 131072  ( 16384)   same K permutation
//   xs_lds  : [96][64]   f32    @ 147456  ( 24576)
//   wih_lds : [512]      f32    @ 172032  (  2048)
//   bsum_lds: [512]      f32    @ 174080  (  2048)
#define XS_BYTE_OFF 147456u
#define SMEM_BYTES (131072 + 16384 + 24576 + 2048 + 2048)

extern __shared__ char smem_raw[];

// CDNA5 hardware tanh (TRANS unit), with safe fallbacks.
#if __has_builtin(__builtin_amdgcn_tanhf)
#define FAST_TANH(x) __builtin_amdgcn_tanhf(x)
#elif __has_builtin(__builtin_amdgcn_tanh_f32)
#define FAST_TANH(x) __builtin_amdgcn_tanh_f32(x)
#else
#define FAST_TANH(x) tanhf(x)
#endif

__device__ __forceinline__ float fast_sig(float x) {
    // sigmoid(x) = 0.5*tanh(0.5x) + 0.5 : 1 TRANS + 2 VALU, branch-free
    return __builtin_fmaf(0.5f, FAST_TANH(0.5f * x), 0.5f);
}

// hidden-index permutation within each 32-block (applied to w_lds K and h_lds)
__device__ __forceinline__ int kperm(int k) {
    return (k & ~31) + ((k & 15) << 1) + ((k >> 4) & 1);
}

__global__ __launch_bounds__(512, 1)
void lstm_point_kernel(const float* __restrict__ x,
                       const float* __restrict__ W_ih,
                       const float* __restrict__ W_hh,
                       const float* __restrict__ b_ih,
                       const float* __restrict__ b_hh,
                       const float* __restrict__ W_fc,
                       const float* __restrict__ b_fc,
                       float* __restrict__ out)
{
    const int p    = blockIdx.x;      // one WG per point
    const int tid  = threadIdx.x;     // 0..511 (16 wave32 waves)
    const int wave = tid >> 5;
    const int lane = tid & 31;
    const int lh   = lane >> 4;       // lane half (0/1)
    const int ln   = lane & 15;

    // wave -> (M tile of batch, hidden-unit quarter)
    const int m = wave & 3;           // batch rows 16m .. 16m+15
    const int q = wave >> 2;          // hidden units 32q .. 32q+31

    __bf16* w_lds    = (__bf16*)smem_raw;            // [512][128]
    __bf16* h_lds    = w_lds + G4 * Hd;              // [64][128]
    float*  xs_lds   = (float*)(h_lds + Bsz * Hd);   // [96][64]
    float*  wih_lds  = xs_lds + Tlen * Bsz;          // [512]
    float*  bsum_lds = wih_lds + G4;                 // [512]

    // ---------------- stage into LDS ----------------
    {
        // xs gather -> LDS via CDNA5 async-to-LDS (ASYNCcnt tracked), GVS mode.
        for (int i = tid; i < Tlen * Bsz; i += 512) {
            const int t = i >> 6, b = i & 63;
            unsigned int ldsoff = XS_BYTE_OFF + (unsigned int)(t * Bsz + b) * 4u;
            unsigned int goff   = (unsigned int)((((b * Tlen) + t) * Pn + p) * 4);
            asm volatile("global_load_async_to_lds_b32 %0, %1, %2"
                         :: "v"(ldsoff), "v"(goff), "s"(x) : "memory");
        }

        // W_hh -> bf16, K columns permuted by kperm
        const float4* whh4 = (const float4*)(W_hh + (size_t)p * G4 * Hd);
        for (int i = tid; i < (G4 * Hd) / 4; i += 512) {
            float4 v = whh4[i];
            const int g = (4 * i) >> 7;       // gate row
            const int k = (4 * i) & 127;      // k base (4 consecutive, same 16-half)
            __bf16* row = w_lds + g * Hd;
            row[kperm(k + 0)] = (__bf16)v.x;
            row[kperm(k + 1)] = (__bf16)v.y;
            row[kperm(k + 2)] = (__bf16)v.z;
            row[kperm(k + 3)] = (__bf16)v.w;
        }
        for (int i = tid; i < G4; i += 512) {
            wih_lds[i]  = W_ih[(size_t)p * G4 + i];
            bsum_lds[i] = b_ih[(size_t)p * G4 + i] + b_hh[(size_t)p * G4 + i];
        }
        for (int i = tid; i < Bsz * Hd; i += 512) h_lds[i] = (__bf16)0.0f;

        asm volatile("s_wait_asynccnt 0x0" ::: "memory");
    }
    __syncthreads();

    // per-lane persistent cell state: creg[st*8+r] for (hu = 32q+16st+ln, b = 16m+8lh+r)
    float  creg[16];
    __bf16 hreg[16];
#pragma unroll
    for (int i = 0; i < 16; ++i) creg[i] = 0.0f;

    // loop-invariant gate coefficients -> registers
    float wi_c[8], bi_c[8];
#pragma unroll
    for (int st = 0; st < 2; ++st) {
        const int hu = q * 32 + st * 16 + ln;
#pragma unroll
        for (int gt = 0; gt < 4; ++gt) {
            wi_c[gt * 2 + st] = wih_lds[gt * Hd + hu];
            bi_c[gt * 2 + st] = bsum_lds[gt * Hd + hu];
        }
    }

    const v8f vzero = {0.f, 0.f, 0.f, 0.f, 0.f, 0.f, 0.f, 0.f};
    const __bf16* aprow = h_lds + (16 * m + ln) * Hd + lh * 16;

    // ---------------- recurrence over T ----------------
    for (int t = 0; t < Tlen; ++t) {
        v8f acc[8];
#pragma unroll
        for (int i = 0; i < 8; ++i) acc[i] = vzero;

        // gates[64,512] = h[64,128] * W_hh^T via 16x16x32 bf16 WMMA
#pragma unroll
        for (int kk = 0; kk < 4; ++kk) {
            const int k0 = kk * 32;
            union { v16bf v; v8bf h[2]; } au;
            au.h[0] = *(const v8bf*)(aprow + k0);
            au.h[1] = *(const v8bf*)(aprow + k0 + 8);

            v16bf barr[8];
#pragma unroll
            for (int nt = 0; nt < 8; ++nt) {
                const int n0 = (nt >> 1) * Hd + q * 32 + (nt & 1) * 16;
                union { v16bf v; v8bf h[2]; } bu;
                const __bf16* bp = w_lds + (n0 + ln) * Hd + k0 + lh * 16;
                bu.h[0] = *(const v8bf*)(bp);
                bu.h[1] = *(const v8bf*)(bp + 8);
                barr[nt] = bu.v;
            }
#pragma unroll
            for (int nt = 0; nt < 8; ++nt) {
                acc[nt] = __builtin_amdgcn_wmma_f32_16x16x32_bf16(
                    false, au.v, false, barr[nt], (short)0, acc[nt], false, false);
            }
        }

#if __has_builtin(__builtin_amdgcn_sched_group_barrier)
        // Software pipeline with half-k-step granularity: ~1.5 k-steps of loads
        // in flight, lower register peak than full 2-step lookahead (avoids the
        // spill seen at 36 outstanding b128 loads). 0x100 = DS-read, 0x008 = WMMA.
        __builtin_amdgcn_sched_group_barrier(0x100, 18, 0); // loads k0 (A+B)
        __builtin_amdgcn_sched_group_barrier(0x100,  9, 0); // loads k1 (1st half)
        __builtin_amdgcn_sched_group_barrier(0x008,  4, 0); // wmma  k0 (1st half)
        __builtin_amdgcn_sched_group_barrier(0x100,  9, 0); // loads k1 (2nd half)
        __builtin_amdgcn_sched_group_barrier(0x008,  4, 0); // wmma  k0 (2nd half)
        __builtin_amdgcn_sched_group_barrier(0x100,  9, 0); // loads k2
        __builtin_amdgcn_sched_group_barrier(0x008,  4, 0); // wmma  k1
        __builtin_amdgcn_sched_group_barrier(0x100,  9, 0); // loads k2
        __builtin_amdgcn_sched_group_barrier(0x008,  4, 0); // wmma  k1
        __builtin_amdgcn_sched_group_barrier(0x100,  9, 0); // loads k3
        __builtin_amdgcn_sched_group_barrier(0x008,  4, 0); // wmma  k2
        __builtin_amdgcn_sched_group_barrier(0x100,  9, 0); // loads k3
        __builtin_amdgcn_sched_group_barrier(0x008,  4, 0); // wmma  k2
        __builtin_amdgcn_sched_group_barrier(0x008,  8, 0); // wmma  k3
#endif

        // Split barrier: signal (our h_lds reads are done), then hide the
        // activation TRANS/VALU work under other waves' arrival.
        asm volatile("s_wait_dscnt 0x0\n\ts_barrier_signal -1" ::: "memory");

        const float* xsrow = xs_lds + t * Bsz;
#pragma unroll
        for (int st = 0; st < 2; ++st) {
#pragma unroll
            for (int r = 0; r < 8; ++r) {
                const int b  = 16 * m + 8 * lh + r;   // C/D: M = 8*(lane/16)+vgpr
                const float xv = xsrow[b];
                float iv = acc[0 + st][r] + __builtin_fmaf(xv, wi_c[0 + st], bi_c[0 + st]);
                float fv = acc[2 + st][r] + __builtin_fmaf(xv, wi_c[2 + st], bi_c[2 + st]);
                float gv = acc[4 + st][r] + __builtin_fmaf(xv, wi_c[4 + st], bi_c[4 + st]);
                float ov = acc[6 + st][r] + __builtin_fmaf(xv, wi_c[6 + st], bi_c[6 + st]);
                iv = fast_sig(iv);
                fv = fast_sig(fv);
                gv = FAST_TANH(gv);
                ov = fast_sig(ov);
                const float c = __builtin_fmaf(fv, creg[st * 8 + r], iv * gv);
                creg[st * 8 + r] = c;
                hreg[st * 8 + r] = (__bf16)(ov * FAST_TANH(c));
            }
        }

        asm volatile("s_barrier_wait -1" ::: "memory");   // everyone's reads done
        // packed h stores: (hu, hu+16) -> adjacent under kperm => one b32/row
        {
            __bf16* hcol = h_lds + q * 32 + ln * 2;
#pragma unroll
            for (int r = 0; r < 8; ++r) {
                const int b = 16 * m + 8 * lh + r;
                v2bf hv = { hreg[r], hreg[8 + r] };
                *(v2bf*)(hcol + b * Hd) = hv;
            }
        }
        __syncthreads();   // h ready for next step / FC
    }

    // ---------------- per-point FC head (un-permute K) ----------------
    {
        const float* wfc = W_fc + (size_t)p * Od * Hd;
        for (int i = tid; i < Bsz * Od; i += 512) {
            const int b  = i / Od;
            const int oo = i % Od;
            const float*  wrow = wfc + oo * Hd;
            const __bf16* hrow = h_lds + b * Hd;
            float s = b_fc[(size_t)p * Od + oo];
#pragma unroll 8
            for (int k = 0; k < Hd; ++k) s += (float)hrow[kperm(k)] * wrow[k];
            out[((size_t)b * Od + oo) * Pn + p] = s;   // [B, O, P]
        }
    }
}

extern "C" void kernel_launch(void* const* d_in, const int* in_sizes, int n_in,
                              void* d_out, int out_size, void* d_ws, size_t ws_size,
                              hipStream_t stream) {
    const float* x    = (const float*)d_in[0];
    const float* W_ih = (const float*)d_in[1];
    const float* W_hh = (const float*)d_in[2];
    const float* b_ih = (const float*)d_in[3];
    const float* b_hh = (const float*)d_in[4];
    const float* W_fc = (const float*)d_in[5];
    const float* b_fc = (const float*)d_in[6];
    float* out = (float*)d_out;

    // 172 KB dynamic LDS per workgroup (CDNA5 WGP has 320 KB)
    (void)hipFuncSetAttribute((const void*)lstm_point_kernel,
                              hipFuncAttributeMaxDynamicSharedMemorySize,
                              (int)SMEM_BYTES);

    lstm_point_kernel<<<dim3(Pn), dim3(512), SMEM_BYTES, stream>>>(
        x, W_ih, W_hh, b_ih, b_hh, W_fc, b_fc, out);
}
